// FrameEntityEncoder_39453569581095
// MI455X (gfx1250) — matrix operand order repaired and checked by reading, not compiled, not run on hardware
//
#include <hip/hip_runtime.h>

// ---------------------------------------------------------------------------
// Fused FrameEntityEncoder for gfx1250 (MI455X), round 3.
// - Weights prepacked (bf16, WMMA-B-fragment swizzled) + fp32 embed-bias table
//   into d_ws -> B loads are 2x global_load_b128, bias is branch-free.
// - 2 batch rows per 256-thread block: each B fragment feeds 2 WMMAs.
// - obs rows staged to LDS via Tensor Data Mover + s_wait_tensorcnt.
// - ALL five matrix products on v_wmma_f32_16x16x32_bf16:
//   embed, QKV, scores QK^T, softmax(P)@V, out-proj.
// - Softmax fully in registers on the WMMA D-fragment (16-lane shuffles).
// ---------------------------------------------------------------------------

typedef __bf16 v16bf __attribute__((ext_vector_type(16)));
typedef __bf16 v8bf  __attribute__((ext_vector_type(8)));
typedef float  v8f   __attribute__((ext_vector_type(8)));
typedef unsigned int u32x4 __attribute__((ext_vector_type(4)));
typedef int    i32x4 __attribute__((ext_vector_type(4)));
typedef int    i32x8 __attribute__((ext_vector_type(8)));

#define EMBED   256
#define SLOTS   16
#define N_ENT   11
#define HEADS   4
#define DH      64
#define OBS_DIM 272
#define KC      128                  // padded combined embed K
#define EBS     (EMBED + 8)          // padded LDS stride (halfs)
#define R       2                    // batch rows per block
#define NFRAG_EMB 64                 // 4 ktiles x 16 ntiles
#define FRAG_Q  64                   // + mat*128 + ktile*16 + ntile
#define FRAG_O  (64 + 3 * 128)
#define NFRAG   (64 + 4 * 128)       // 576 fragments, 1KB each; bias table after

#define WMMA_BF16(A, Bv, C) \
  __builtin_amdgcn_wmma_f32_16x16x32_bf16(false, (A), false, (Bv), (short)0, (C), false, false)

// ---- A fragment: 16x32 bf16 tile from LDS (lane half h: K=16h..16h+15) -----
__device__ __forceinline__ v16bf load_A_lds(const __bf16* base, int stride, int kbase) {
  const int lane = threadIdx.x & 31;
  const int m    = lane & 15;
  const int kh   = (lane >> 4) << 4;           // 0 or 16
  const __bf16* r = base + m * stride + kbase + kh;
  v8bf lo = *(const v8bf*)(r);
  v8bf hi = *(const v8bf*)(r + 8);
  v16bf a;
#pragma unroll
  for (int i = 0; i < 8; ++i) { a[i] = lo[i]; a[8 + i] = hi[i]; }
  return a;
}

// ---- B fragment from LDS where B[k][n] is stored as arr[n][k] (k contig) ---
// lanes 0-15: K=kbase..+15 of column nbase+lane; lanes 16-31: K=kbase+16..+31.
__device__ __forceinline__ v16bf load_B_lds_T(const __bf16* base, int stride,
                                              int nbase, int kbase) {
  const int lane = threadIdx.x & 31;
  const int n  = nbase + (lane & 15);
  const int k0 = kbase + ((lane >> 4) << 4);
  const __bf16* r = base + n * stride + k0;
  v8bf lo = *(const v8bf*)(r);
  v8bf hi = *(const v8bf*)(r + 8);
  v16bf b;
#pragma unroll
  for (int i = 0; i < 8; ++i) { b[i] = lo[i]; b[8 + i] = hi[i]; }
  return b;
}

// ---- B fragment: prepacked, lane-contiguous (32 bytes per lane) ------------
__device__ __forceinline__ v16bf load_B_ws(const __bf16* wsbuf, int frag) {
  const int lane = threadIdx.x & 31;
  return *(const v16bf*)(wsbuf + ((size_t)frag << 9) + (lane << 4));
}

// ---- Virtual concatenated embed weight [Ws;Wp;We;pad;Wm;pad] (128x256) -----
__device__ __forceinline__ float wcomb(const float* Ws, const float* Wp,
                                       const float* We, const float* Wm,
                                       int k, int n) {
  if (k < 32)  return Ws[k * EMBED + n];
  if (k < 64)  return Wp[(k - 32) * EMBED + n];
  if (k < 88)  return We[(k - 64) * EMBED + n];
  if (k < 96)  return 0.f;
  if (k < 112) return Wm[(k - 96) * EMBED + n];
  return 0.f;
}

// ---------------------------------------------------------------------------
// Prepack: weights -> bf16 B-fragments; embed bias -> fp32 table (16x256).
// ---------------------------------------------------------------------------
__global__ __launch_bounds__(256)
void prepack_weights(const float* __restrict__ Ws, const float* __restrict__ Wp,
                     const float* __restrict__ We, const float* __restrict__ Wm,
                     const float* __restrict__ Wq, const float* __restrict__ Wk,
                     const float* __restrict__ Wv, const float* __restrict__ Wo,
                     const float* __restrict__ bs, const float* __restrict__ bp,
                     const float* __restrict__ be, const float* __restrict__ bm,
                     __bf16* __restrict__ wsbuf, float* __restrict__ biasE) {
  const int f = blockIdx.x;
  if (f < NFRAG) {
    for (int e = threadIdx.x; e < 512; e += 256) {
      const int l = e >> 4, i = e & 15;
      const int kit = ((l >> 4) << 4) + i;   // 0..31 within K-tile
      const int nit = l & 15;                // 0..15 within N-tile
      float v;
      if (f < NFRAG_EMB) {
        const int ktile = f >> 4, ntile = f & 15;
        v = wcomb(Ws, Wp, We, Wm, ktile * 32 + kit, ntile * 16 + nit);
      } else {
        const int g = f - NFRAG_EMB;
        const int mi = g >> 7, rem = g & 127;
        const int ktile = rem >> 4, ntile = rem & 15;
        const float* W = (mi == 0) ? Wq : (mi == 1) ? Wk : (mi == 2) ? Wv : Wo;
        v = W[(ktile * 32 + kit) * EMBED + ntile * 16 + nit];
      }
      wsbuf[(size_t)f * 512 + e] = (__bf16)v;
    }
  } else {
    const int slot = f - NFRAG;            // 0..15
    const int n = threadIdx.x;
    const float v = (slot == 0) ? bs[n] : (slot == 1) ? bp[n]
                  : (slot <= 9) ? be[n] : (slot == 10) ? bm[n] : 0.f;
    biasE[slot * EMBED + n] = v;
  }
}

// ---------------------------------------------------------------------------
// Main fused kernel: 2 rows / block, 256 threads (8 wave32s).
// ---------------------------------------------------------------------------
__global__ __launch_bounds__(256)
void frame_entity_encoder_fused(
    const float* __restrict__ obs,
    const float* __restrict__ bq, const float* __restrict__ bk,
    const float* __restrict__ bv, const float* __restrict__ bo,
    const float* __restrict__ ln_g, const float* __restrict__ ln_b,
    const __bf16* __restrict__ wsbuf, const float* __restrict__ biasE,
    float* __restrict__ out) {

  __shared__ __align__(16) float  sObs[R * OBS_DIM];          // TDM landing pad
  __shared__ __align__(16) __bf16 sA[R][SLOTS][KC];           // embed A (block-diag)
  __shared__ __align__(16) __bf16 sEb[R][SLOTS][EBS];         // entities -> residual
  __shared__ __align__(16) __bf16 sQ[R][SLOTS][EBS];
  __shared__ __align__(16) __bf16 sK[R][SLOTS][EBS];
  __shared__ __align__(16) __bf16 sVt[R][EMBED][32];          // V^T [col][j], j>=16 zero
  __shared__ __align__(16) __bf16 sP[R][HEADS][SLOTS][32];    // softmax probs, A-tile form
  __shared__ __align__(16) __bf16 sAO[R][SLOTS][EBS];         // attn out -> normed

  const int tid  = threadIdx.x;
  const int lane = tid & 31;
  const int wave = tid >> 5;
  const int row0 = blockIdx.x * R;
  const float* orow0 = obs + (size_t)row0 * OBS_DIM;

  __builtin_prefetch(wsbuf + (size_t)tid * 512, 0, 2);        // warm L2 on weights

  // ---------- Stage 0: TDM async copy of both obs rows into LDS -------------
  if (wave == 0) {
    const unsigned lds_off = (unsigned)(uintptr_t)(const void*)&sObs[0];
    const unsigned long long ga = (unsigned long long)(uintptr_t)orow0;
    u32x4 g0 = {};
    g0[0] = 1u;                                   // count=1 (valid user D#)
    g0[1] = lds_off;                              // lds_addr
    g0[2] = (unsigned)(ga & 0xFFFFFFFFu);         // global_addr[31:0]
    g0[3] = (unsigned)((ga >> 32) & 0x01FFFFFFu)  // global_addr[56:32]
          | (2u << 30);                           // type=2 ("image")
    i32x8 g1 = {};
    g1[0] = 0x00020000;                           // data_size=4B, no multicast
    g1[1] = (int)((unsigned)(R * OBS_DIM) << 16); // tensor_dim0 = 544
    g1[2] = (int)(1u << 16);                      // tensor_dim1 = 1
    g1[3] = (int)((unsigned)(R * OBS_DIM) << 16); // tile_dim0 = 544
    g1[4] = 1;                                    // tile_dim1 = 1
    g1[5] = R * OBS_DIM;                          // tensor_dim0_stride
    i32x4 z4 = {};
#if __has_include(<hip/amd_detail/amd_gfx1250_TDM.h>)
    i32x8 z8 = {};
    __builtin_amdgcn_tensor_load_to_lds(g0, g1, z4, z4, z8, 0);
#else
    __builtin_amdgcn_tensor_load_to_lds(g0, g1, z4, z4, 0);
#endif
    __builtin_amdgcn_s_wait_tensorcnt((short)0);
  }
  __syncthreads();

  // ---------- Stage 1: scatter obs into block-diag A; zero V^T pad rows -----
  {
    __bf16* flat = &sA[0][0][0];
    for (int idx = tid; idx < R * SLOTS * KC; idx += 256) {
      const int r2 = idx >> 11, rem = idx & 2047;
      const int slot = rem >> 7, col = rem & (KC - 1);
      const float* o = sObs + r2 * OBS_DIM;
      float v = 0.f;
      if (slot == 0)      { if (col < 32)               v = o[col]; }
      else if (slot == 1) { if (col >= 32 && col < 64)  v = o[col]; }
      else if (slot < 10) { if (col >= 64 && col < 88)  v = o[64 + (slot - 2) * 24 + (col - 64)]; }
      else if (slot == 10){ if (col >= 96 && col < 112) v = o[256 + (col - 96)]; }
      flat[idx] = (__bf16)v;
    }
    for (int idx = tid; idx < R * EMBED * 16; idx += 256) {   // sVt[.][.][16..31] = 0
      const int r2 = idx >> 12, rem = idx & 4095;
      sVt[r2][rem >> 4][(rem & 15) + 16] = (__bf16)0.f;
    }
  }
  __syncthreads();

  // ---------- Stage 2: embed GEMM (16x128 @ 128x256) + bias + relu ----------
#pragma unroll
  for (int tt = 0; tt < 2; ++tt) {
    const int nt = wave * 2 + tt, nbase = nt << 4;
    v8f acc0 = {}, acc1 = {};
#pragma unroll
    for (int kk = 0; kk < KC; kk += 32) {
      const v16bf b = load_B_ws(wsbuf, (kk >> 5) * 16 + nt);
      const v16bf a0 = load_A_lds(&sA[0][0][0], KC, kk);
      const v16bf a1 = load_A_lds(&sA[1][0][0], KC, kk);
      acc0 = WMMA_BF16(a0, b, acc0);
      acc1 = WMMA_BF16(a1, b, acc1);
    }
    const int n = nbase + (lane & 15);
    const int mh = (lane >> 4) << 3;
#pragma unroll
    for (int r = 0; r < 8; ++r) {
      const int m = r + mh;
      const float bias = biasE[m * EMBED + n];   // rows >= 11 are zero
      sEb[0][m][n] = (__bf16)fmaxf(acc0[r] + bias, 0.f);
      sEb[1][m][n] = (__bf16)fmaxf(acc1[r] + bias, 0.f);
    }
  }
  __syncthreads();

  // ---------- Stage 3: Q/K/V GEMMs (48 N-tile tasks over 8 waves) -----------
  for (int t = wave; t < 48; t += 8) {
    const int mat = t >> 4;                       // 0=Q 1=K 2=V
    const int nt = t & 15, nbase = nt << 4;
    v8f acc0 = {}, acc1 = {};
#pragma unroll
    for (int kk = 0; kk < EMBED; kk += 32) {
      const v16bf b = load_B_ws(wsbuf, FRAG_Q + mat * 128 + (kk >> 5) * 16 + nt);
      const v16bf a0 = load_A_lds(&sEb[0][0][0], EBS, kk);
      const v16bf a1 = load_A_lds(&sEb[1][0][0], EBS, kk);
      acc0 = WMMA_BF16(a0, b, acc0);
      acc1 = WMMA_BF16(a1, b, acc1);
    }
    const int n = nbase + (lane & 15);
    const int mh = (lane >> 4) << 3;
    const float* bsel = (mat == 0) ? bq : (mat == 1) ? bk : bv;
    const float bias = bsel[n];
    const float scale = (mat == 0) ? 0.125f : 1.f; // 1/sqrt(Dh) folded into Q
    if (mat == 2) {
#pragma unroll
      for (int r = 0; r < 8; ++r) {
        const int m = r + mh;
        sVt[0][n][m] = (__bf16)(acc0[r] + bias);  // store V transposed
        sVt[1][n][m] = (__bf16)(acc1[r] + bias);
      }
    } else {
      __bf16 (*d0)[EBS] = (mat == 0) ? sQ[0] : sK[0];
      __bf16 (*d1)[EBS] = (mat == 0) ? sQ[1] : sK[1];
#pragma unroll
      for (int r = 0; r < 8; ++r) {
        const int m = r + mh;
        d0[m][n] = (__bf16)((acc0[r] + bias) * scale);
        d1[m][n] = (__bf16)((acc1[r] + bias) * scale);
      }
    }
  }
  __syncthreads();

  // ---------- Stage 4: scores = Q K^T via WMMA; softmax in registers --------
  {
    const int r2 = wave >> 2, h = wave & 3;       // one (row, head) per wave
    v8f acc = {};
#pragma unroll
    for (int kk = 0; kk < DH; kk += 32) {
      const v16bf a = load_A_lds(&sQ[r2][0][0], EBS, h * DH + kk);
      const v16bf b = load_B_lds_T(&sK[r2][0][0], EBS, 0, h * DH + kk);
      acc = WMMA_BF16(a, b, acc);
    }
    const int j  = lane & 15;                     // lane -> column j
    const int mh = (lane >> 4) << 3;
    const bool valid = (j < N_ENT);
#pragma unroll
    for (int r = 0; r < 8; ++r) {
      const int i = r + mh;
      const float x = valid ? acc[r] : -3.0e38f;
      float mx = x;
#pragma unroll
      for (int off = 8; off > 0; off >>= 1)
        mx = fmaxf(mx, __shfl_xor(mx, off, 16));
      const float e = valid ? __expf(x - mx) : 0.f;
      float den = e;
#pragma unroll
      for (int off = 8; off > 0; off >>= 1)
        den += __shfl_xor(den, off, 16);
      const float p = e / den;
      sP[r2][h][i][j]      = (__bf16)p;           // zero-padded 16x32 A-tile
      sP[r2][h][i][j + 16] = (__bf16)0.f;
    }
  }
  __syncthreads();

  // ---------- Stage 5: attn-out = P @ V via WMMA (32 tasks / 8 waves) -------
  for (int t = wave; t < 32; t += 8) {
    const int r2 = t >> 4, h = (t >> 2) & 3, n4 = t & 3;
    const int nbase = h * DH + n4 * 16;
    const v16bf a = load_A_lds(&sP[r2][h][0][0], 32, 0);
    const v16bf b = load_B_lds_T(&sVt[r2][0][0], 32, nbase, 0);
    v8f acc = {};
    acc = WMMA_BF16(a, b, acc);
    const int c = nbase + (lane & 15);
    const int mh = (lane >> 4) << 3;
#pragma unroll
    for (int r = 0; r < 8; ++r)
      sAO[r2][r + mh][c] = (__bf16)acc[r];
  }
  __syncthreads();

  // ---------- Stage 6: out-proj GEMM + bias + residual (into sEb) -----------
#pragma unroll
  for (int tt = 0; tt < 2; ++tt) {
    const int nt = wave * 2 + tt, nbase = nt << 4;
    v8f acc0 = {}, acc1 = {};
#pragma unroll
    for (int kk = 0; kk < EMBED; kk += 32) {
      const v16bf b = load_B_ws(wsbuf, FRAG_O + (kk >> 5) * 16 + nt);
      const v16bf a0 = load_A_lds(&sAO[0][0][0], EBS, kk);
      const v16bf a1 = load_A_lds(&sAO[1][0][0], EBS, kk);
      acc0 = WMMA_BF16(a0, b, acc0);
      acc1 = WMMA_BF16(a1, b, acc1);
    }
    const int n = nbase + (lane & 15);
    const int mh = (lane >> 4) << 3;
#pragma unroll
    for (int r = 0; r < 8; ++r) {
      const int m = r + mh;
      const float bias = bo[n];
      sEb[0][m][n] = (__bf16)((float)sEb[0][m][n] + acc0[r] + bias);
      sEb[1][m][n] = (__bf16)((float)sEb[1][m][n] + acc1[r] + bias);
    }
  }
  __syncthreads();

  // ---------- Stage 7: LayerNorm per entity (wave per slot, shuffles) -------
  for (int idx = wave; idx < R * N_ENT; idx += 8) {
    const int r2 = (idx < N_ENT) ? 0 : 1;
    const int m  = idx - r2 * N_ENT;
    float s = 0.f, s2 = 0.f;
#pragma unroll
    for (int k = 0; k < 8; ++k) {
      const float x = (float)sEb[r2][m][lane + 32 * k];
      s += x; s2 += x * x;
    }
#pragma unroll
    for (int off = 16; off > 0; off >>= 1) {
      s  += __shfl_xor(s,  off, 32);
      s2 += __shfl_xor(s2, off, 32);
    }
    const float mu   = s * (1.f / 256.f);
    const float var  = s2 * (1.f / 256.f) - mu * mu;
    const float rstd = rsqrtf(var + 1e-5f);
#pragma unroll
    for (int k = 0; k < 8; ++k) {
      const int n = lane + 32 * k;
      const float x = (float)sEb[r2][m][n];
      sAO[r2][m][n] = (__bf16)((x - mu) * rstd * ln_g[n] + ln_b[n]);
    }
  }
  __syncthreads();

  // ---------- Stage 8: mean over the 11 entities, store ---------------------
  for (int idx = tid; idx < R * EMBED; idx += 256) {
    const int r2 = idx >> 8, n = idx & 255;
    float s = 0.f;
#pragma unroll
    for (int m = 0; m < N_ENT; ++m) s += (float)sAO[r2][m][n];
    out[(size_t)(row0 + r2) * EMBED + n] = s * (1.f / (float)N_ENT);
  }
}

extern "C" void kernel_launch(void* const* d_in, const int* in_sizes, int n_in,
                              void* d_out, int out_size, void* d_ws, size_t ws_size,
                              hipStream_t stream) {
  (void)in_sizes; (void)n_in; (void)ws_size; (void)out_size;
  const float* obs = (const float*)d_in[0];
  const float* Ws  = (const float*)d_in[1];
  const float* bs  = (const float*)d_in[2];
  const float* Wp  = (const float*)d_in[3];
  const float* bp  = (const float*)d_in[4];
  const float* We  = (const float*)d_in[5];
  const float* be  = (const float*)d_in[6];
  const float* Wm  = (const float*)d_in[7];
  const float* bm  = (const float*)d_in[8];
  const float* Wq  = (const float*)d_in[9];
  const float* bq  = (const float*)d_in[10];
  const float* Wk  = (const float*)d_in[11];
  const float* bk  = (const float*)d_in[12];
  const float* Wv  = (const float*)d_in[13];
  const float* bv  = (const float*)d_in[14];
  const float* Wo  = (const float*)d_in[15];
  const float* bo  = (const float*)d_in[16];
  const float* ln_g = (const float*)d_in[17];
  const float* ln_b = (const float*)d_in[18];
  float* outp = (float*)d_out;

  __bf16* wsbuf = (__bf16*)d_ws;                       // 576KB fragments
  float*  biasE = (float*)(wsbuf + (size_t)NFRAG * 512); // +16KB bias table

  prepack_weights<<<NFRAG + 16, 256, 0, stream>>>(
      Ws, Wp, We, Wm, Wq, Wk, Wv, Wo, bs, bp, be, bm, wsbuf, biasE);

  const int B = 32768;
  frame_entity_encoder_fused<<<B / R, 256, 0, stream>>>(
      obs, bq, bk, bv, bo, ln_g, ln_b, wsbuf, biasE, outp);
}
